// TokenEncoder_61684320305428
// MI455X (gfx1250) — matrix operand order repaired and checked by reading, not compiled, not run on hardware
//
#include <hip/hip_runtime.h>
#include <hip/hip_bf16.h>
#include <stdint.h>

// Problem constants (match reference)
#define BB 8
#define LL 2048
#define DD 512
#define DMM 1024
#define SS 32
#define PP 4096

typedef __attribute__((ext_vector_type(16))) __bf16 v16bf;
typedef __attribute__((ext_vector_type(8)))  __bf16 v8bf;
typedef __attribute__((ext_vector_type(8)))  float  v8f;
typedef __attribute__((ext_vector_type(4)))  float  v4f;

// ---------------------------------------------------------------------------
// Phase 0: zero counters (workspace is poisoned; must not rely on prior state)
// ---------------------------------------------------------------------------
__global__ void init_kernel(int* __restrict__ ws) {
    if (threadIdx.x < 64) ws[threadIdx.x] = 0;   // counts[32] | fill[32]
}

// ---------------------------------------------------------------------------
// Phase 1: histogram of sid over valid (non-padded) tokens
// ---------------------------------------------------------------------------
__global__ void count_kernel(const int* __restrict__ sid,
                             const unsigned char* __restrict__ mask,
                             int* __restrict__ counts) {
    int t = blockIdx.x * blockDim.x + threadIdx.x;
    if (t < BB * LL && mask[t]) atomicAdd(&counts[sid[t]], 1);
}

// ---------------------------------------------------------------------------
// Phase 2: exclusive prefix sum over S=32 buckets
// ---------------------------------------------------------------------------
__global__ void scan_kernel(const int* __restrict__ counts, int* __restrict__ offs) {
    if (threadIdx.x == 0 && blockIdx.x == 0) {
        int acc = 0;
        for (int s = 0; s < SS; ++s) { offs[s] = acc; acc += counts[s]; }
        offs[SS] = acc;
    }
}

// ---------------------------------------------------------------------------
// Phase 3: scatter flat token indices into per-sid buckets
// ---------------------------------------------------------------------------
__global__ void scatter_kernel(const int* __restrict__ sid,
                               const unsigned char* __restrict__ mask,
                               const int* __restrict__ offs,
                               int* __restrict__ fill,
                               int* __restrict__ tok) {
    int t = blockIdx.x * blockDim.x + threadIdx.x;
    if (t >= BB * LL) return;
    if (!mask[t]) return;
    int s = sid[t];
    int p = atomicAdd(&fill[s], 1);
    tok[offs[s] + p] = t;
}

// ---------------------------------------------------------------------------
// Phase 3b: split fp32 -> (bf16 hi, bf16 lo) planes, 8 elements/thread
// ---------------------------------------------------------------------------
__global__ void cvt_split_kernel(const float* __restrict__ in,
                                 unsigned short* __restrict__ hip_,
                                 unsigned short* __restrict__ lop_,
                                 int n) {
    int i = (blockIdx.x * blockDim.x + threadIdx.x) * 8;
    if (i >= n) return;
    __bf16* hi = (__bf16*)hip_;
    __bf16* lo = (__bf16*)lop_;
    v4f x0 = *(const v4f*)(in + i);
    v4f x1 = *(const v4f*)(in + i + 4);
    v8bf h, l;
    #pragma unroll
    for (int j = 0; j < 4; ++j) {
        float a = x0[j];
        __bf16 ha = (__bf16)a;
        h[j] = ha; l[j] = (__bf16)(a - (float)ha);
        float b = x1[j];
        __bf16 hb = (__bf16)b;
        h[j + 4] = hb; l[j + 4] = (__bf16)(b - (float)hb);
    }
    *(v8bf*)(hi + i) = h;
    *(v8bf*)(lo + i) = l;
}

// ---------------------------------------------------------------------------
// Phase 4: initialize output with embeddings + bias + CLS row; write attn_keep
// ---------------------------------------------------------------------------
__global__ void embed_kernel(const int* __restrict__ pos,
                             const int* __restrict__ sid,
                             const int* __restrict__ mod,
                             const int* __restrict__ role,
                             const unsigned char* __restrict__ mask,
                             const float* __restrict__ b_proj,
                             const float* __restrict__ cls_content,
                             const float* __restrict__ pos_emb,
                             const float* __restrict__ id_emb,
                             const float* __restrict__ mod_emb,
                             const float* __restrict__ role_emb,
                             float* __restrict__ out) {
    int row = blockIdx.x;                 // 0 .. B*(L+1)-1
    int b = row / (LL + 1);
    int l = row - b * (LL + 1);
    int c = threadIdx.x * 4;              // 256 threads * 4 = DM

    v4f v;
    float keep;
    if (l == 0) {
        v4f cc = *(const v4f*)(cls_content + c);
        v4f pe = *(const v4f*)(pos_emb + c);                   // pos 0
        v4f ie = *(const v4f*)(id_emb + (size_t)SS * DMM + c); // cls id = S
        v = cc + pe + ie;
        keep = 1.0f;
    } else {
        int t = b * LL + (l - 1);
        int p = pos[t], s = sid[t], m = mod[t], r = role[t];
        bool valid = mask[t] != 0;
        v4f pe = *(const v4f*)(pos_emb  + (size_t)p * DMM + c);
        v4f ie = *(const v4f*)(id_emb   + (size_t)s * DMM + c);
        v4f me = *(const v4f*)(mod_emb  + (size_t)m * DMM + c);
        v4f re = *(const v4f*)(role_emb + (size_t)r * DMM + c);
        v = pe + ie + me + re;
        if (valid) v = v + *(const v4f*)(b_proj + (size_t)s * DMM + c);
        keep = valid ? 1.0f : 0.0f;
    }
    *(v4f*)(out + (size_t)row * DMM + c) = v;
    if (threadIdx.x == 0)
        out[(size_t)BB * (LL + 1) * DMM + row] = keep;
}

// ---------------------------------------------------------------------------
// Shared helper: map flat M-tile id -> (bucket, tile-in-bucket), gather rows
// ---------------------------------------------------------------------------
__device__ __forceinline__ bool
tile_setup(const int* __restrict__ offs, const int* __restrict__ tok,
           int* s_meta, int* s_row, int& sig_out) {
    const int tid = threadIdx.x;
    if (tid == 0) {
        int t = blockIdx.x, s = 0;
        for (; s < SS; ++s) {
            int cnt = offs[s + 1] - offs[s];
            int nt = (cnt + 15) >> 4;
            if (t < nt) break;
            t -= nt;
        }
        s_meta[0] = s; s_meta[1] = t;
    }
    __syncthreads();
    const int sig = s_meta[0];
    sig_out = sig;
    if (sig >= SS) return false;           // tile beyond total work
    const int mt   = s_meta[1];
    const int boff = offs[sig];
    const int cnt  = offs[sig + 1] - boff;
    if (tid < 16) {
        int i = mt * 16 + tid;
        s_row[tid] = (i < cnt) ? tok[boff + i] : -1;
    }
    __syncthreads();
    return true;
}

// ---------------------------------------------------------------------------
// Phase 5 (fast path): bucketed MoE GEMM on preconverted bf16 hi/lo planes.
//   grid.x = flat M-tile, grid.y = 256-column group; 8 waves/block,
//   each wave computes TWO 16x16 N-tiles (cols n0 and n0+128) reusing A.
//   fp32-accurate product = hi*hi + hi*lo + lo*hi (3 WMMAs per B tile).
// ---------------------------------------------------------------------------
__global__ void __launch_bounds__(256)
moe_gemm_pre_kernel(const unsigned short* __restrict__ ehi_u,
                    const unsigned short* __restrict__ elo_u,
                    const unsigned short* __restrict__ whi_u,
                    const unsigned short* __restrict__ wlo_u,
                    const int* __restrict__ offs,
                    const int* __restrict__ tok,
                    float* __restrict__ out) {
    __shared__ int s_meta[2];
    __shared__ int s_row[16];
    int sig;
    if (!tile_setup(offs, tok, s_meta, s_row, sig)) return;

    const __bf16* ehi = (const __bf16*)ehi_u;
    const __bf16* elo = (const __bf16*)elo_u;
    const __bf16* whi = (const __bf16*)whi_u;
    const __bf16* wlo = (const __bf16*)wlo_u;

    const int lane    = threadIdx.x & 31;
    const int wave    = threadIdx.x >> 5;
    const int nl      = lane & 15;
    const int hi_half = lane >> 4;
    const int n0      = blockIdx.y * 256 + wave * 16;   // tile 0
    const int n1      = n0 + 128;                       // tile 1

    const int arow_idx = s_row[nl];
    const size_t abase = (size_t)(arow_idx < 0 ? 0 : arow_idx) * DD
                       + (hi_half ? 16 : 0);            // aoff
    const __bf16* ah = ehi + abase;
    const __bf16* al = elo + abase;

    const size_t b0base = ((size_t)sig * DMM + n0 + nl) * DD + (hi_half ? 16 : 0);
    const size_t b1base = ((size_t)sig * DMM + n1 + nl) * DD + (hi_half ? 16 : 0);
    const __bf16* b0h = whi + b0base;
    const __bf16* b0l = wlo + b0base;
    const __bf16* b1h = whi + b1base;
    const __bf16* b1l = wlo + b1base;

    v8f acc0 = {}, acc1 = {};
    #pragma unroll 2
    for (int k0 = 0; k0 < DD; k0 += 32) {
        // A: e0..15 = K [k0+aoff,+16)
        v8bf ah0 = *(const v8bf*)(ah + k0);
        v8bf ah1 = *(const v8bf*)(ah + k0 + 8);
        v8bf al0 = *(const v8bf*)(al + k0);
        v8bf al1 = *(const v8bf*)(al + k0 + 8);
        v16bf Ahi = __builtin_shufflevector(ah0, ah1,
                      0,1,2,3,4,5,6,7,8,9,10,11,12,13,14,15);
        v16bf Alo = __builtin_shufflevector(al0, al1,
                      0,1,2,3,4,5,6,7,8,9,10,11,12,13,14,15);
        // B: 16 consecutive K values per lane
        v16bf B0h = *(const v16bf*)(b0h + k0);
        v16bf B0l = *(const v16bf*)(b0l + k0);
        acc0 = __builtin_amdgcn_wmma_f32_16x16x32_bf16(false, Ahi, false, B0h,
                                                       (short)0, acc0, false, false);
        acc0 = __builtin_amdgcn_wmma_f32_16x16x32_bf16(false, Ahi, false, B0l,
                                                       (short)0, acc0, false, false);
        acc0 = __builtin_amdgcn_wmma_f32_16x16x32_bf16(false, Alo, false, B0h,
                                                       (short)0, acc0, false, false);
        v16bf B1h = *(const v16bf*)(b1h + k0);
        v16bf B1l = *(const v16bf*)(b1l + k0);
        acc1 = __builtin_amdgcn_wmma_f32_16x16x32_bf16(false, Ahi, false, B1h,
                                                       (short)0, acc1, false, false);
        acc1 = __builtin_amdgcn_wmma_f32_16x16x32_bf16(false, Ahi, false, B1l,
                                                       (short)0, acc1, false, false);
        acc1 = __builtin_amdgcn_wmma_f32_16x16x32_bf16(false, Alo, false, B1h,
                                                       (short)0, acc1, false, false);
    }

    // C/D layout: VGPR r -> M = r + 8*hi_half, N = lane&15
    #pragma unroll
    for (int r = 0; r < 8; ++r) {
        int m = r + (hi_half ? 8 : 0);
        int trow = s_row[m];
        if (trow >= 0) {
            int b = trow / LL;                           // out row = trow + b + 1
            size_t o = (size_t)(trow + b + 1) * DMM;
            out[o + n0 + nl] += acc0[r];
            out[o + n1 + nl] += acc1[r];
        }
    }
}

// ---------------------------------------------------------------------------
// Phase 5 (fallback path, small workspace): in-loop split-bf16 GEMM
// ---------------------------------------------------------------------------
__global__ void __launch_bounds__(256)
moe_gemm_kernel(const float* __restrict__ emb,
                const float* __restrict__ Wp,
                const int* __restrict__ offs,
                const int* __restrict__ tok,
                float* __restrict__ out) {
    __shared__ int s_meta[2];
    __shared__ int s_row[16];
    int sig;
    if (!tile_setup(offs, tok, s_meta, s_row, sig)) return;

    const int lane    = threadIdx.x & 31;
    const int wave    = threadIdx.x >> 5;
    const int nl      = lane & 15;
    const int hi_half = lane >> 4;
    const int n0      = blockIdx.y * 128 + wave * 16;

    const int   arow_idx = s_row[nl];
    const float* arow = emb + (size_t)(arow_idx < 0 ? 0 : arow_idx) * DD;
    const float* brow = Wp + ((size_t)sig * DMM + n0 + nl) * DD + (hi_half ? 16 : 0);
    const int aoff = hi_half ? 16 : 0;

    v8f acc = {};
    #pragma unroll 4
    for (int k0 = 0; k0 < DD; k0 += 32) {
        v4f a0 = *(const v4f*)(arow + k0 + aoff);
        v4f a1 = *(const v4f*)(arow + k0 + aoff + 4);
        v4f a2 = *(const v4f*)(arow + k0 + aoff + 8);
        v4f a3 = *(const v4f*)(arow + k0 + aoff + 12);
        v4f b0 = *(const v4f*)(brow + k0);
        v4f b1 = *(const v4f*)(brow + k0 + 4);
        v4f b2 = *(const v4f*)(brow + k0 + 8);
        v4f b3 = *(const v4f*)(brow + k0 + 12);

        float av[16], bv[16];
        #pragma unroll
        for (int j = 0; j < 4; ++j) {
            av[j]      = a0[j]; av[j + 4]  = a1[j];
            av[j + 8]  = a2[j]; av[j + 12] = a3[j];
            bv[j]      = b0[j]; bv[j + 4]  = b1[j];
            bv[j + 8]  = b2[j]; bv[j + 12] = b3[j];
        }
        v16bf ahi, alo, bhi, blo;
        #pragma unroll
        for (int e = 0; e < 16; ++e) {
            float xa = av[e];
            __bf16 ha = (__bf16)xa;
            ahi[e] = ha; alo[e] = (__bf16)(xa - (float)ha);
            float xb = bv[e];
            __bf16 hb = (__bf16)xb;
            bhi[e] = hb; blo[e] = (__bf16)(xb - (float)hb);
        }
        acc = __builtin_amdgcn_wmma_f32_16x16x32_bf16(false, ahi, false, bhi,
                                                      (short)0, acc, false, false);
        acc = __builtin_amdgcn_wmma_f32_16x16x32_bf16(false, ahi, false, blo,
                                                      (short)0, acc, false, false);
        acc = __builtin_amdgcn_wmma_f32_16x16x32_bf16(false, alo, false, bhi,
                                                      (short)0, acc, false, false);
    }
    const int col = n0 + nl;
    #pragma unroll
    for (int r = 0; r < 8; ++r) {
        int m = r + (hi_half ? 8 : 0);
        int trow = s_row[m];
        if (trow >= 0) {
            int b = trow / LL;
            out[(size_t)(trow + b + 1) * DMM + col] += acc[r];
        }
    }
}

// ---------------------------------------------------------------------------
extern "C" void kernel_launch(void* const* d_in, const int* in_sizes, int n_in,
                              void* d_out, int out_size, void* d_ws, size_t ws_size,
                              hipStream_t stream) {
    const float*         emb       = (const float*)d_in[0];
    const int*           pos       = (const int*)d_in[1];
    const int*           sid       = (const int*)d_in[2];
    const int*           mod       = (const int*)d_in[3];
    const int*           role      = (const int*)d_in[4];
    const unsigned char* mask      = (const unsigned char*)d_in[5];
    const float*         W_proj    = (const float*)d_in[6];
    const float*         b_proj    = (const float*)d_in[7];
    const float*         cls       = (const float*)d_in[8];
    const float*         pos_emb   = (const float*)d_in[9];
    const float*         id_emb    = (const float*)d_in[10];
    const float*         mod_emb   = (const float*)d_in[11];
    const float*         role_emb  = (const float*)d_in[12];
    float*               out       = (float*)d_out;

    // Workspace: [0,256KB) int scratch | ehi | elo | whi | wlo  (bf16 planes)
    int* ws      = (int*)d_ws;
    int* counts  = ws;
    int* fill    = ws + 32;
    int* offs    = ws + 64;
    int* tok     = ws + 128;

    const int    NTOK   = BB * LL;                          // 16384
    const int    MT_MAX = NTOK / 16 + SS;                   // 1056 M-tiles max
    const int    NE     = BB * LL * DD;                     // 8,388,608
    const int    NW     = SS * DMM * DD;                    // 16,777,216
    const size_t INTRES = 256 * 1024;
    unsigned short* ehi = (unsigned short*)((char*)d_ws + INTRES);
    unsigned short* elo = ehi + (size_t)NE;
    unsigned short* whi = elo + (size_t)NE;
    unsigned short* wlo = whi + (size_t)NW;
    const size_t need = INTRES + 2u * sizeof(unsigned short) * ((size_t)NE + (size_t)NW);

    init_kernel<<<1, 64, 0, stream>>>(ws);
    count_kernel<<<(NTOK + 255) / 256, 256, 0, stream>>>(sid, mask, counts);
    scan_kernel<<<1, 32, 0, stream>>>(counts, offs);
    scatter_kernel<<<(NTOK + 255) / 256, 256, 0, stream>>>(sid, mask, offs, fill, tok);
    embed_kernel<<<BB * (LL + 1), 256, 0, stream>>>(pos, sid, mod, role, mask,
                                                    b_proj, cls, pos_emb, id_emb,
                                                    mod_emb, role_emb, out);
    if (ws_size >= need) {
        cvt_split_kernel<<<NE / 8 / 256, 256, 0, stream>>>(emb, ehi, elo, NE);
        cvt_split_kernel<<<NW / 8 / 256, 256, 0, stream>>>(W_proj, whi, wlo, NW);
        moe_gemm_pre_kernel<<<dim3(MT_MAX, DMM / 256), 256, 0, stream>>>(
            ehi, elo, whi, wlo, offs, tok, out);
    } else {
        moe_gemm_kernel<<<dim3(MT_MAX, DMM / 128), 256, 0, stream>>>(
            emb, W_proj, offs, tok, out);
    }
}
